// DiTBlock_43954695307440
// MI455X (gfx1250) — matrix operand order repaired and checked by reading, not compiled, not run on hardware
//
#include <hip/hip_runtime.h>
#include <hip/hip_bf16.h>

typedef __attribute__((ext_vector_type(16))) __bf16 v16bf;
typedef __attribute__((ext_vector_type(8)))  __bf16 v8bf;
typedef __attribute__((ext_vector_type(8)))  float  v8f;

union Frag { v16bf v; v8bf h2[2]; };

#define HH    1024
#define NHEAD 16
#define BB    2
#define NN    2048
#define MMF   1024      // past-frame tokens
#define SSL   128       // language tokens
#define LKCA  1184      // 1024+128+1 = 1153 padded to multiple of 32
#define MLPH  4096
#define SIXH  6144

static __device__ __forceinline__ v8f zero8() {
  v8f z;
  for (int i = 0; i < 8; ++i) z[i] = 0.f;
  return z;
}

// ---------------------------------------------------------------------------
// elementwise / small kernels
// ---------------------------------------------------------------------------

// fp32 (K,N) row-major -> bf16 (N,K) row-major (weight transpose)
__global__ void k_wt_bf16(const float* __restrict__ w, __bf16* __restrict__ wt,
                          int K, int Nn) {
  size_t i = (size_t)blockIdx.x * blockDim.x + threadIdx.x;
  size_t total = (size_t)K * Nn;
  if (i >= total) return;
  int n = (int)(i % Nn);
  int k = (int)(i / Nn);
  wt[(size_t)n * K + k] = (__bf16)w[i];
}

__global__ void k_f32_to_bf16(const float* __restrict__ a, __bf16* __restrict__ o, size_t n) {
  size_t i = (size_t)blockIdx.x * blockDim.x + threadIdx.x;
  if (i < n) o[i] = (__bf16)a[i];
}

// one wave per output element: y[b,j] = act(dot(x[b,:], w[:,j]) + bias[j])
__global__ void k_gemv(const float* __restrict__ xin, const float* __restrict__ w,
                       const float* __restrict__ bias, float* __restrict__ yout,
                       int K, int Nout, int do_silu) {
  int wv = blockIdx.x * (blockDim.x >> 5) + (threadIdx.x >> 5);
  if (wv >= BB * Nout) return;
  int lane = threadIdx.x & 31;
  int b = wv / Nout, j = wv % Nout;
  float acc = 0.f;
  for (int k = lane; k < K; k += 32)
    acc += xin[(size_t)b * K + k] * w[(size_t)k * Nout + j];
  for (int m = 16; m > 0; m >>= 1) acc += __shfl_xor(acc, m);
  if (lane == 0) {
    float v = acc + bias[j];
    if (do_silu) v = v / (1.f + __expf(-v));
    yout[(size_t)b * Nout + j] = v;
  }
}

// LayerNorm over H=1024, optional affine (gamma/beta) and AdaLN modulate, bf16 out.
// One block (256 thr) per row; rows = B*N.
__global__ void k_ln_mod(const float* __restrict__ x, __bf16* __restrict__ y,
                         const float* __restrict__ gamma, const float* __restrict__ beta,
                         const float* __restrict__ mods, int shift_off, int scale_off) {
  int row = blockIdx.x;
  int b = row / NN;
  const float* xr = x + (size_t)row * HH;
  float s = 0.f, s2 = 0.f;
  for (int c = threadIdx.x; c < HH; c += blockDim.x) { float v = xr[c]; s += v; s2 += v * v; }
  for (int m = 16; m > 0; m >>= 1) { s += __shfl_xor(s, m); s2 += __shfl_xor(s2, m); }
  __shared__ float rs[8], rs2[8], smean, sinv;
  int wid = threadIdx.x >> 5, lane = threadIdx.x & 31;
  if (lane == 0) { rs[wid] = s; rs2[wid] = s2; }
  __syncthreads();
  if (threadIdx.x == 0) {
    float a = 0.f, q = 0.f;
    for (int i = 0; i < (int)(blockDim.x >> 5); ++i) { a += rs[i]; q += rs2[i]; }
    float mean = a * (1.f / HH);
    float var  = q * (1.f / HH) - mean * mean;
    smean = mean; sinv = rsqrtf(var + 1e-6f);
  }
  __syncthreads();
  float mean = smean, inv = sinv;
  __bf16* yr = y + (size_t)row * HH;
  for (int c = threadIdx.x; c < HH; c += blockDim.x) {
    float v = (xr[c] - mean) * inv;
    if (gamma) v = v * gamma[c] + beta[c];
    if (mods)  v = v * (1.f + mods[(size_t)b * SIXH + scale_off + c])
                  + mods[(size_t)b * SIXH + shift_off + c];
    yr[c] = (__bf16)v;
  }
}

// 3D-RoPE + head split + bf16 pack: src fp32 (B,Nr,cst) at col offset co
// -> dst bf16 (B,NHEAD,Lout,64) at row offset ro. One thread per rotation pair.
__global__ void k_rope_pack(const float* __restrict__ src, int co, int cst,
                            __bf16* __restrict__ dst, int Lout, int ro,
                            const int* __restrict__ tid_, const int* __restrict__ hid_,
                            const int* __restrict__ wid_, int Nr) {
  size_t idx = (size_t)blockIdx.x * blockDim.x + threadIdx.x;
  size_t total = (size_t)BB * Nr * NHEAD * 32;
  if (idx >= total) return;
  int p  = (int)(idx & 31);
  size_t rest = idx >> 5;
  int hd = (int)(rest & (NHEAD - 1));
  size_t bn = rest >> 4;
  int n = (int)(bn % Nr);
  int b = (int)(bn / Nr);
  int base, hs, i, dim;
  const int* ids;
  if (p < 8)       { base = 0;  hs = 8;  i = p;      dim = 16; ids = tid_; }
  else if (p < 16) { base = 16; hs = 8;  i = p - 8;  dim = 16; ids = hid_; }
  else             { base = 32; hs = 16; i = p - 16; dim = 32; ids = wid_; }
  const float* sp = src + (size_t)(b * Nr + n) * cst + co + hd * 64;
  float x1 = sp[base + i], x2 = sp[base + hs + i];
  float o1 = x1, o2 = x2;
  if (ids) {
    float fr  = __expf(-9.210340372f * (float)(2 * i) / (float)dim); // 10000^{-2i/dim}
    float ang = (float)ids[n] * fr;
    float cc = __cosf(ang), ssn = __sinf(ang);
    o1 = x1 * cc - x2 * ssn;
    o2 = x2 * cc + x1 * ssn;
  }
  __bf16* d = dst + ((size_t)(b * NHEAD + hd) * Lout + ro + n) * 64;
  d[base + i]      = (__bf16)o1;
  d[base + hs + i] = (__bf16)o2;
}

// V pack transposed: src fp32 (B,Nr,cst)+co -> dst bf16 (B,NHEAD,64,Lout)+col ro
__global__ void k_pack_vt(const float* __restrict__ src, int co, int cst,
                          __bf16* __restrict__ dst, int Lout, int ro, int Nr) {
  size_t idx = (size_t)blockIdx.x * blockDim.x + threadIdx.x;
  size_t total = (size_t)BB * Nr * NHEAD * 64;
  if (idx >= total) return;
  int d  = (int)(idx & 63);
  size_t rest = idx >> 6;
  int hd = (int)(rest & (NHEAD - 1));
  size_t bn = rest >> 4;
  int n = (int)(bn % Nr);
  int b = (int)(bn / Nr);
  dst[((size_t)(b * NHEAD + hd) * 64 + d) * Lout + ro + n] =
      (__bf16)src[(size_t)(b * Nr + n) * cst + co + hd * 64 + d];
}

// additive attention mask over padded CA key axis
__global__ void k_mask(const unsigned char* __restrict__ lm, float* __restrict__ mask) {
  int i = blockIdx.x * blockDim.x + threadIdx.x;
  if (i >= BB * LKCA) return;
  int b = i / LKCA, k = i % LKCA;
  float v = 0.f;
  if (k >= MMF + SSL + 1) v = -1e30f;                       // padding keys
  else if (k >= MMF && k < MMF + SSL) {
    if (lm[b * SSL + (k - MMF)]) v = -1e30f;                // ignored language keys
  }
  mask[i] = v;
}

__global__ void k_gelu_bf16(const float* __restrict__ a, __bf16* __restrict__ o, size_t n) {
  size_t i = (size_t)blockIdx.x * blockDim.x + threadIdx.x;
  if (i >= n) return;
  float x = a[i];
  float g = 0.5f * x * (1.f + tanhf(0.7978845608f * (x + 0.044715f * x * x * x)));
  o[i] = (__bf16)g;
}

// x += gate * delta (gate from mods at goff, or 1.0 if goff<0)
__global__ void k_addgate(float* __restrict__ x, const float* __restrict__ dlt,
                          const float* __restrict__ mods, int goff) {
  size_t i = (size_t)blockIdx.x * blockDim.x + threadIdx.x;
  if (i >= (size_t)BB * NN * HH) return;
  int c = (int)(i % HH);
  int b = (int)(i / ((size_t)NN * HH));
  float g = (goff >= 0) ? mods[(size_t)b * SIXH + goff + c] : 1.f;
  x[i] += g * dlt[i];
}

__global__ void k_final(const float* __restrict__ x, const float* __restrict__ dlt,
                        const float* __restrict__ mods, int goff, float* __restrict__ outp) {
  size_t i = (size_t)blockIdx.x * blockDim.x + threadIdx.x;
  if (i >= (size_t)BB * NN * HH) return;
  int c = (int)(i % HH);
  int b = (int)(i / ((size_t)NN * HH));
  outp[i] = x[i] + mods[(size_t)b * SIXH + goff + c] * dlt[i];
}

// ---------------------------------------------------------------------------
// WMMA bf16 GEMM:  C(M,N) fp32 = A(M,K) bf16 @ Wt(N,K)^T bf16 + bias
// Register-blocked: each wave computes a 32x64 C tile (2 A frags x 4 B frags,
// 8 v_wmma_f32_16x16x32_bf16 per K-step of 32 -> 12 b128 loads per 8 WMMAs).
// K is a template constant (1024/4096) so row strides fold to shifts/immediate
// offsets and the address pointers stay in registers (no scratch spills).
// Requires N % 64 == 0; M handled with row clamp + guarded stores.
// ---------------------------------------------------------------------------
template <int KK>
__global__ void k_gemm(const __bf16* __restrict__ A, const __bf16* __restrict__ Wt,
                       const float* __restrict__ bias, float* __restrict__ Cc,
                       int Mt, int Nn) {
  int wv = blockIdx.x * (blockDim.x >> 5) + (threadIdx.x >> 5);
  int ntiles = Nn >> 6;                 // 64-wide tiles
  int mtiles = (Mt + 31) >> 5;          // 32-high tiles
  if (wv >= mtiles * ntiles) return;
  int tm = wv / ntiles, tn = wv % ntiles;
  int lane = threadIdx.x & 31, half = lane >> 4, l16 = lane & 15;

  int ar0 = tm * 32 + l16;       if (ar0 >= Mt) ar0 = Mt - 1;   // clamp, stores guarded
  int ar1 = tm * 32 + 16 + l16;  if (ar1 >= Mt) ar1 = Mt - 1;
  const __bf16* ap0 = A + (size_t)ar0 * KK + half * 8;
  const __bf16* ap1 = A + (size_t)ar1 * KK + half * 8;
  const __bf16* bp0 = Wt + (size_t)(tn * 64 + l16) * KK + half * 8;

  v8f acc[2][4];
#pragma unroll
  for (int mi = 0; mi < 2; ++mi)
#pragma unroll
    for (int t = 0; t < 4; ++t) acc[mi][t] = zero8();

  for (int k0 = 0; k0 < KK; k0 += 32) {
    __builtin_prefetch((const void*)(ap0 + k0 + 256), 0, 1);
    __builtin_prefetch((const void*)(bp0 + k0 + 256), 0, 1);
    Frag fa0, fa1, fb[4];
    fa0.h2[0] = *(const v8bf*)(ap0 + k0);                   // K 0..7  / 8..15
    fa0.h2[1] = *(const v8bf*)(ap0 + k0 + 16);              // K 16..23/ 24..31
    fa1.h2[0] = *(const v8bf*)(ap1 + k0);
    fa1.h2[1] = *(const v8bf*)(ap1 + k0 + 16);
#pragma unroll
    for (int t = 0; t < 4; ++t) {
      const __bf16* bp = bp0 + (size_t)t * 16 * KK + k0;    // constant-folded stride
      fb[t].h2[0] = *(const v8bf*)(bp);                     // K 0..7  / 8..15
      fb[t].h2[1] = *(const v8bf*)(bp + 16);                // K 16..23/ 24..31
    }
#pragma unroll
    for (int t = 0; t < 4; ++t)
      acc[0][t] = __builtin_amdgcn_wmma_f32_16x16x32_bf16(false, fa0.v, false, fb[t].v,
                                                          (short)0, acc[0][t], false, false);
#pragma unroll
    for (int t = 0; t < 4; ++t)
      acc[1][t] = __builtin_amdgcn_wmma_f32_16x16x32_bf16(false, fa1.v, false, fb[t].v,
                                                          (short)0, acc[1][t], false, false);
  }
#pragma unroll
  for (int mi = 0; mi < 2; ++mi) {
    int rbase = tm * 32 + mi * 16 + half * 8;
#pragma unroll
    for (int t = 0; t < 4; ++t) {
      int col = tn * 64 + t * 16 + l16;
      float bb = bias ? bias[col] : 0.f;
#pragma unroll
      for (int j = 0; j < 8; ++j) {
        int r = rbase + j;
        if (r < Mt) Cc[(size_t)r * Nn + col] = acc[mi][t][j] + bb;
      }
    }
  }
}

// ---------------------------------------------------------------------------
// Flash attention (wave32): 16 queries/wave, 32-key chunks, WMMA QK^T + PV,
// online softmax with half-wave shuffles, LDS C->A layout transpose for P.
// All K/V fragment loads for a chunk are hoisted ahead of their WMMA group so
// the scheduler can clause them and stage s_wait_loadcnt instead of full waits.
// Q: (B,NHEAD,Nq,64) bf16   K: (B,NHEAD,Lk,64) bf16   Vt: (B,NHEAD,64,Lk) bf16
// out: (B,Nq,H) bf16 merged heads.  mask: (B,Lk) additive fp32 or null.
// ---------------------------------------------------------------------------
__global__ void k_flash(const __bf16* __restrict__ Q, const __bf16* __restrict__ Kb,
                        const __bf16* __restrict__ Vt, const float* __restrict__ mask,
                        __bf16* __restrict__ outp, int Nq, int Lk, float scale) {
  int wid = threadIdx.x >> 5, lane = threadIdx.x & 31;
  int half = lane >> 4, l16 = lane & 15;
  int wv = blockIdx.x * (blockDim.x >> 5) + wid;
  int qtiles = Nq >> 4;
  if (wv >= BB * NHEAD * qtiles) return;
  int qt = wv % qtiles, bh = wv / qtiles;
  int b = bh / NHEAD, hh = bh % NHEAD;

  const __bf16* Qp = Q + ((size_t)bh * Nq + qt * 16 + l16) * 64;
  const __bf16* Kp = Kb + (size_t)bh * Lk * 64;
  const __bf16* Vp = Vt + (size_t)bh * 64 * Lk;

  Frag qf[2];
#pragma unroll
  for (int f = 0; f < 2; ++f) {
    qf[f].h2[0] = *(const v8bf*)(Qp + f * 32 + half * 8);
    qf[f].h2[1] = *(const v8bf*)(Qp + f * 32 + half * 8 + 16);
  }

  float m[8], ls[8];
  v8f o[4];
#pragma unroll
  for (int j = 0; j < 8; ++j) { m[j] = -1e30f; ls[j] = 0.f; }
#pragma unroll
  for (int t = 0; t < 4; ++t) o[t] = zero8();

  // per-wave P staging tile, row stride 40 ushorts (80B) = conflict-free b128 reads
  __shared__ __attribute__((aligned(16))) __bf16 sP[4][16 * 40];
  __bf16* sp = sP[wid];
  const float* mrow = mask ? (mask + (size_t)b * Lk) : (const float*)0;

  for (int kb = 0; kb < Lk; kb += 32) {
    // ---- load all 4 K fragments for this 32-key chunk up front ----
    const __bf16* kp0 = Kp + (size_t)(kb + l16) * 64 + half * 8;
    const __bf16* kp1 = Kp + (size_t)(kb + 16 + l16) * 64 + half * 8;
    Frag kf[4];
    kf[0].h2[0] = *(const v8bf*)(kp0);
    kf[0].h2[1] = *(const v8bf*)(kp0 + 16);
    kf[1].h2[0] = *(const v8bf*)(kp0 + 32);
    kf[1].h2[1] = *(const v8bf*)(kp0 + 48);
    kf[2].h2[0] = *(const v8bf*)(kp1);
    kf[2].h2[1] = *(const v8bf*)(kp1 + 16);
    kf[3].h2[0] = *(const v8bf*)(kp1 + 32);
    kf[3].h2[1] = *(const v8bf*)(kp1 + 48);
    float ma0 = mrow ? mrow[kb + l16] : 0.f;
    float ma1 = mrow ? mrow[kb + 16 + l16] : 0.f;

    // ---- S = Q K^T for 32 keys (two 16-key groups) ----
    v8f s0 = zero8(), s1 = zero8();
    s0 = __builtin_amdgcn_wmma_f32_16x16x32_bf16(false, qf[0].v, false, kf[0].v, (short)0, s0, false, false);
    s0 = __builtin_amdgcn_wmma_f32_16x16x32_bf16(false, qf[1].v, false, kf[1].v, (short)0, s0, false, false);
    s1 = __builtin_amdgcn_wmma_f32_16x16x32_bf16(false, qf[0].v, false, kf[2].v, (short)0, s1, false, false);
    s1 = __builtin_amdgcn_wmma_f32_16x16x32_bf16(false, qf[1].v, false, kf[3].v, (short)0, s1, false, false);
#pragma unroll
    for (int j = 0; j < 8; ++j) {
      s0[j] = s0[j] * scale + ma0;
      s1[j] = s1[j] * scale + ma1;
    }
    // ---- online softmax (row stats live in 16-lane halves) ----
    float mn[8];
#pragma unroll
    for (int j = 0; j < 8; ++j) mn[j] = fmaxf(s0[j], s1[j]);
#pragma unroll
    for (int msk = 1; msk < 16; msk <<= 1)
#pragma unroll
      for (int j = 0; j < 8; ++j) mn[j] = fmaxf(mn[j], __shfl_xor(mn[j], msk));
    float al[8], ps[8];
#pragma unroll
    for (int j = 0; j < 8; ++j) {
      float mnew = fmaxf(m[j], mn[j]);
      al[j] = __expf(m[j] - mnew);
      m[j] = mnew;
    }
#pragma unroll
    for (int j = 0; j < 8; ++j) {
      float p0 = __expf(s0[j] - m[j]);
      float p1 = __expf(s1[j] - m[j]);
      sp[(j + half * 8) * 40 + l16]      = (__bf16)p0;
      sp[(j + half * 8) * 40 + 16 + l16] = (__bf16)p1;
      ps[j] = p0 + p1;
    }
    // ---- load all 4 V fragments while softmax reductions proceed ----
    Frag vf[4];
#pragma unroll
    for (int t = 0; t < 4; ++t) {
      const __bf16* vp = Vp + (size_t)(t * 16 + l16) * Lk + kb + half * 8;
      vf[t].h2[0] = *(const v8bf*)vp;
      vf[t].h2[1] = *(const v8bf*)(vp + 16);
    }
#pragma unroll
    for (int msk = 1; msk < 16; msk <<= 1)
#pragma unroll
      for (int j = 0; j < 8; ++j) ps[j] += __shfl_xor(ps[j], msk);
#pragma unroll
    for (int j = 0; j < 8; ++j) ls[j] = ls[j] * al[j] + ps[j];
#pragma unroll
    for (int t = 0; t < 4; ++t)
#pragma unroll
      for (int j = 0; j < 8; ++j) o[t][j] *= al[j];
    // ---- read P back in A-matrix layout (same-wave LDS ops are in-order) ----
    Frag pf_;
    {
      const __bf16* pr = sp + l16 * 40 + half * 8;
      pf_.h2[0] = *(const v8bf*)(pr);
      pf_.h2[1] = *(const v8bf*)(pr + 16);
    }
    // ---- O += P @ V for 4 groups of 16 dims ----
#pragma unroll
    for (int t = 0; t < 4; ++t)
      o[t] = __builtin_amdgcn_wmma_f32_16x16x32_bf16(false, pf_.v, false, vf[t].v,
                                                     (short)0, o[t], false, false);
  }
  // ---- epilogue: normalize and store merged-heads bf16 ----
#pragma unroll
  for (int t = 0; t < 4; ++t)
#pragma unroll
    for (int j = 0; j < 8; ++j) {
      int q = qt * 16 + j + half * 8;
      float val = o[t][j] / ls[j];
      outp[((size_t)b * Nq + q) * HH + hh * 64 + t * 16 + l16] = (__bf16)val;
    }
}

// ---------------------------------------------------------------------------
// host
// ---------------------------------------------------------------------------
static inline unsigned ew_blocks(size_t n) { return (unsigned)((n + 255) / 256); }

extern "C" void kernel_launch(void* const* d_in, const int* in_sizes, int n_in,
                              void* d_out, int out_size, void* d_ws, size_t ws_size,
                              hipStream_t stream) {
  (void)in_sizes; (void)n_in; (void)out_size; (void)ws_size;
  const float* x   = (const float*)d_in[0];
  const float* tt  = (const float*)d_in[1];
  const float* pf  = (const float*)d_in[2];
  const float* lx  = (const float*)d_in[3];
  const float* stk = (const float*)d_in[4];
  const int* xt = (const int*)d_in[5];
  const int* xh = (const int*)d_in[6];
  const int* xw = (const int*)d_in[7];
  const int* pt = (const int*)d_in[8];
  const int* ph = (const int*)d_in[9];
  const int* pw = (const int*)d_in[10];
  const unsigned char* lmask = (const unsigned char*)d_in[11];
  const float* adaln_w1 = (const float*)d_in[12];
  const float* adaln_b1 = (const float*)d_in[13];
  const float* adaln_w2 = (const float*)d_in[14];
  const float* adaln_b2 = (const float*)d_in[15];
  const float* sa_qkv_w = (const float*)d_in[16];
  const float* sa_qkv_b = (const float*)d_in[17];
  const float* sa_out_w = (const float*)d_in[18];
  const float* sa_out_b = (const float*)d_in[19];
  const float* nca_w    = (const float*)d_in[20];
  const float* nca_b    = (const float*)d_in[21];
  const float* ca_q_w   = (const float*)d_in[22];
  const float* ca_q_b   = (const float*)d_in[23];
  const float* ca_k_w   = (const float*)d_in[24];
  const float* ca_k_b   = (const float*)d_in[25];
  const float* ca_v_w   = (const float*)d_in[26];
  const float* ca_v_b   = (const float*)d_in[27];
  const float* ca_out_w = (const float*)d_in[28];
  const float* ca_out_b = (const float*)d_in[29];
  const float* mlp_w1   = (const float*)d_in[30];
  const float* mlp_b1   = (const float*)d_in[31];
  const float* mlp_w2   = (const float*)d_in[32];
  const float* mlp_b2   = (const float*)d_in[33];

  // deterministic workspace arena
  char* ws = (char*)d_ws;
  size_t off = 0;
  auto alloc = [&](size_t bytes) -> char* {
    char* p = ws + off;
    off = (off + bytes + 255) & ~(size_t)255;
    return p;
  };
  __bf16* wt_qkv   = (__bf16*)alloc((size_t)3072 * 1024 * 2);
  __bf16* wt_saout = (__bf16*)alloc((size_t)1024 * 1024 * 2);
  __bf16* wt_caq   = (__bf16*)alloc((size_t)1024 * 1024 * 2);
  __bf16* wt_cak   = (__bf16*)alloc((size_t)3 * 1024 * 1024 * 2);
  __bf16* wt_cav   = (__bf16*)alloc((size_t)3 * 1024 * 1024 * 2);
  __bf16* wt_caout = (__bf16*)alloc((size_t)1024 * 1024 * 2);
  __bf16* wt_mlp1  = (__bf16*)alloc((size_t)4096 * 1024 * 2);
  __bf16* wt_mlp2  = (__bf16*)alloc((size_t)4096 * 1024 * 2);
  float*  hbuf = (float*)alloc((size_t)BB * HH * 4);
  float*  mods = (float*)alloc((size_t)BB * SIXH * 4);
  __bf16* xs   = (__bf16*)alloc((size_t)BB * NN * HH * 2);
  float*  gbuf = (float*)alloc((size_t)4096 * 4096 * 4);        // fp32 GEMM results
  char*   region1 = alloc((size_t)4096 * 4096 * 2);             // 32MB, time-shared
  __bf16* attnb = (__bf16*)alloc((size_t)BB * NN * HH * 2);
  float*  obuf  = (float*)alloc((size_t)BB * NN * HH * 4);
  float*  xcur  = (float*)alloc((size_t)BB * NN * HH * 4);
  __bf16* pfb = (__bf16*)alloc((size_t)BB * MMF * HH * 2);
  __bf16* lb  = (__bf16*)alloc((size_t)BB * SSL * HH * 2);
  __bf16* stb = (__bf16*)alloc((size_t)BB * 1 * HH * 2);
  float*  maskb = (float*)alloc((size_t)BB * LKCA * 4);

  const size_t slot = (size_t)8 * 1024 * 1024;                  // 8MB sub-slots
  __bf16* Qsa = (__bf16*)(region1);
  __bf16* Ksa = (__bf16*)(region1 + slot);
  __bf16* Vtsa = (__bf16*)(region1 + 2 * slot);
  __bf16* Qca = Qsa;                                            // reuse after SA
  __bf16* Kca = Ksa;
  __bf16* Vtca = Vtsa;
  __bf16* midb = (__bf16*)region1;                              // reuse in MLP

  dim3 tb(256);

  // ---- residual state ----
  hipMemcpyAsync(xcur, x, (size_t)BB * NN * HH * 4, hipMemcpyDeviceToDevice, stream);

  // ---- weights -> transposed bf16 ----
  auto wt = [&](const float* src, __bf16* dst, int K, int Nn) {
    k_wt_bf16<<<ew_blocks((size_t)K * Nn), tb, 0, stream>>>(src, dst, K, Nn);
  };
  wt(sa_qkv_w, wt_qkv, 1024, 3072);
  wt(sa_out_w, wt_saout, 1024, 1024);
  wt(ca_q_w, wt_caq, 1024, 1024);
  for (int i = 0; i < 3; ++i) {
    wt(ca_k_w + (size_t)i * 1024 * 1024, wt_cak + (size_t)i * 1024 * 1024, 1024, 1024);
    wt(ca_v_w + (size_t)i * 1024 * 1024, wt_cav + (size_t)i * 1024 * 1024, 1024, 1024);
  }
  wt(ca_out_w, wt_caout, 1024, 1024);
  wt(mlp_w1, wt_mlp1, 1024, 4096);
  wt(mlp_w2, wt_mlp2, 4096, 1024);

  // ---- CA sources -> bf16 ----
  k_f32_to_bf16<<<ew_blocks((size_t)BB * MMF * HH), tb, 0, stream>>>(pf, pfb, (size_t)BB * MMF * HH);
  k_f32_to_bf16<<<ew_blocks((size_t)BB * SSL * HH), tb, 0, stream>>>(lx, lb, (size_t)BB * SSL * HH);
  k_f32_to_bf16<<<ew_blocks((size_t)BB * HH), tb, 0, stream>>>(stk, stb, (size_t)BB * HH);

  // ---- AdaLN ----
  k_gemv<<<(BB * HH + 3) / 4, dim3(128), 0, stream>>>(tt, adaln_w1, adaln_b1, hbuf, HH, HH, 1);
  k_gemv<<<(BB * SIXH + 3) / 4, dim3(128), 0, stream>>>(hbuf, adaln_w2, adaln_b2, mods, HH, SIXH, 0);

  auto gemm = [&](const __bf16* A, const __bf16* W, const float* bias, float* C,
                  int M, int Nn, int K) {
    int waves = ((M + 31) / 32) * (Nn / 64);    // 32x64 C tile per wave
    unsigned blocks = (unsigned)((waves + 3) / 4);
    if (K == 1024)
      k_gemm<1024><<<blocks, dim3(128), 0, stream>>>(A, W, bias, C, M, Nn);
    else
      k_gemm<4096><<<blocks, dim3(128), 0, stream>>>(A, W, bias, C, M, Nn);
  };

  // ================= Self-attention =================
  k_ln_mod<<<BB * NN, tb, 0, stream>>>(xcur, xs, nullptr, nullptr, mods, 0, 1024);
  gemm(xs, wt_qkv, sa_qkv_b, gbuf, BB * NN, 3072, 1024);
  {
    size_t nthr = (size_t)BB * NN * NHEAD * 32;
    k_rope_pack<<<ew_blocks(nthr), tb, 0, stream>>>(gbuf, 0, 3072, Qsa, NN, 0, xt, xh, xw, NN);
    k_rope_pack<<<ew_blocks(nthr), tb, 0, stream>>>(gbuf, 1024, 3072, Ksa, NN, 0, xt, xh, xw, NN);
    size_t nv = (size_t)BB * NN * HH;
    k_pack_vt<<<ew_blocks(nv), tb, 0, stream>>>(gbuf, 2048, 3072, Vtsa, NN, 0, NN);
  }
  {
    int waves = BB * NHEAD * (NN / 16);
    k_flash<<<(waves + 3) / 4, dim3(128), 0, stream>>>(Qsa, Ksa, Vtsa, nullptr, attnb,
                                                       NN, NN, 0.125f);
  }
  gemm(attnb, wt_saout, sa_out_b, obuf, BB * NN, 1024, 1024);
  k_addgate<<<ew_blocks((size_t)BB * NN * HH), tb, 0, stream>>>(xcur, obuf, mods, 2048);

  // ================= Cross-attention =================
  k_ln_mod<<<BB * NN, tb, 0, stream>>>(xcur, xs, nca_w, nca_b, nullptr, 0, 0);
  gemm(xs, wt_caq, ca_q_b, gbuf, BB * NN, 1024, 1024);
  k_rope_pack<<<ew_blocks((size_t)BB * NN * NHEAD * 32), tb, 0, stream>>>(
      gbuf, 0, 1024, Qca, NN, 0, xt, xh, xw, NN);

  hipMemsetAsync(Kca, 0, slot, stream);
  hipMemsetAsync(Vtca, 0, slot, stream);
  // source 0: past frames (RoPE on K)
  gemm(pfb, wt_cak, ca_k_b, gbuf, BB * MMF, 1024, 1024);
  k_rope_pack<<<ew_blocks((size_t)BB * MMF * NHEAD * 32), tb, 0, stream>>>(
      gbuf, 0, 1024, Kca, LKCA, 0, pt, ph, pw, MMF);
  gemm(pfb, wt_cav, ca_v_b, gbuf, BB * MMF, 1024, 1024);
  k_pack_vt<<<ew_blocks((size_t)BB * MMF * HH), tb, 0, stream>>>(gbuf, 0, 1024, Vtca, LKCA, 0, MMF);
  // source 1: language tokens
  gemm(lb, wt_cak + (size_t)1024 * 1024, ca_k_b + 1024, gbuf, BB * SSL, 1024, 1024);
  k_rope_pack<<<ew_blocks((size_t)BB * SSL * NHEAD * 32), tb, 0, stream>>>(
      gbuf, 0, 1024, Kca, LKCA, MMF, nullptr, nullptr, nullptr, SSL);
  gemm(lb, wt_cav + (size_t)1024 * 1024, ca_v_b + 1024, gbuf, BB * SSL, 1024, 1024);
  k_pack_vt<<<ew_blocks((size_t)BB * SSL * HH), tb, 0, stream>>>(gbuf, 0, 1024, Vtca, LKCA, MMF, SSL);
  // source 2: state token
  gemm(stb, wt_cak + (size_t)2 * 1024 * 1024, ca_k_b + 2048, gbuf, BB * 1, 1024, 1024);
  k_rope_pack<<<ew_blocks((size_t)BB * 1 * NHEAD * 32), tb, 0, stream>>>(
      gbuf, 0, 1024, Kca, LKCA, MMF + SSL, nullptr, nullptr, nullptr, 1);
  gemm(stb, wt_cav + (size_t)2 * 1024 * 1024, ca_v_b + 2048, gbuf, BB * 1, 1024, 1024);
  k_pack_vt<<<ew_blocks((size_t)BB * 1 * HH), tb, 0, stream>>>(gbuf, 0, 1024, Vtca, LKCA, MMF + SSL, 1);

  k_mask<<<ew_blocks((size_t)BB * LKCA), tb, 0, stream>>>(lmask, maskb);
  {
    int waves = BB * NHEAD * (NN / 16);
    k_flash<<<(waves + 3) / 4, dim3(128), 0, stream>>>(Qca, Kca, Vtca, maskb, attnb,
                                                       NN, LKCA, 0.125f);
  }
  gemm(attnb, wt_caout, ca_out_b, obuf, BB * NN, 1024, 1024);
  k_addgate<<<ew_blocks((size_t)BB * NN * HH), tb, 0, stream>>>(xcur, obuf, mods, -1);

  // ================= MLP =================
  k_ln_mod<<<BB * NN, tb, 0, stream>>>(xcur, xs, nullptr, nullptr, mods, 3072, 4096);
  gemm(xs, wt_mlp1, mlp_b1, gbuf, BB * NN, MLPH, 1024);
  k_gelu_bf16<<<ew_blocks((size_t)BB * NN * MLPH), tb, 0, stream>>>(gbuf, midb, (size_t)BB * NN * MLPH);
  gemm(midb, wt_mlp2, mlp_b2, obuf, BB * NN, 1024, MLPH);
  k_final<<<ew_blocks((size_t)BB * NN * HH), tb, 0, stream>>>(xcur, obuf, mods, 5120, (float*)d_out);
}
